// A2CDense_90512140796485
// MI455X (gfx1250) — matrix operand order repaired and checked by reading, not compiled, not run on hardware
//
#include <hip/hip_runtime.h>
#include <hip/hip_bf16.h>
#include <stdint.h>

typedef __attribute__((ext_vector_type(16))) _Float16 v16h;
typedef __attribute__((ext_vector_type(8)))  _Float16 v8h;
typedef __attribute__((ext_vector_type(8)))  float    v8f;

#define ACT_STRIDE 128
#define MLP_WAVES  4

union V16U { v16h v; v8h h[2]; };

// ---------------------------------------------------------------------------
// WMMA fragment loaders (CDNA5 16x16x32 f16 layouts, ISA 7.12.2)
// A: lanes 0-15 rows M=0..15 with K = {kt..kt+7, kt+16..kt+23};
//    lanes 16-31 same rows, K = {kt+8..kt+15, kt+24..kt+31}
//    -> two contiguous 16B runs per lane.
// ---------------------------------------------------------------------------
__device__ __forceinline__ v16h load_afrag(const _Float16* act, int lane, int kt) {
  int r  = lane & 15;
  int kb = kt + ((lane & 16) ? 8 : 0);
  const _Float16* p = act + r * ACT_STRIDE + kb;
  V16U a;
  a.h[0] = *(const v8h*)(p);        // K = kb + 0..7
  a.h[1] = *(const v8h*)(p + 16);   // K = kb + 16..23
  return a.v;
}

// B fragments are staged in LDS pre-swizzled: frag-major, 32 lanes x 16 halves
// contiguous, so each lane reads one 32-byte run.
__device__ __forceinline__ v16h load_bfrag_sw(const _Float16* wl, int ntiles,
                                              int lane, int ktIdx, int nt) {
  const _Float16* p = wl + ((ktIdx * ntiles + nt) << 9) + (lane << 4);
  return *(const v16h*)p;
}

// stage f32 weights [Kraw,C] -> f16 LDS in fragment-swizzled layout (zero-pad K)
__device__ __forceinline__ void stage_w_sw(_Float16* dst, const float* W,
                                           int Kraw, int K, int C, int tid) {
  int ntiles = C >> 4;
  for (int i = tid; i < K * C; i += 128) {
    int k = i / C, c = i - k * C;
    float v = (k < Kraw) ? W[(size_t)k * C + c] : 0.f;
    int frag = (k >> 5) * ntiles + (c >> 4);
    int lane = (c & 15) + ((k & 8) ? 16 : 0);
    int j    = (k & 7) + ((k & 16) ? 8 : 0);
    dst[(frag << 9) + (lane << 4) + j] = (_Float16)v;
  }
}

// hidden layer: C = 64, relu, writes back into the wave's activation tile
__device__ __forceinline__ void layer64(_Float16* act, const _Float16* wl,
                                        const float* bias, int K, int lane) {
  v8f a0 = {}, a1 = {}, a2 = {}, a3 = {};
  int nk = K >> 5;
  for (int ki = 0; ki < nk; ki++) {
    v16h af = load_afrag(act, lane, ki << 5);
    v16h b0 = load_bfrag_sw(wl, 4, lane, ki, 0);
    a0 = __builtin_amdgcn_wmma_f32_16x16x32_f16(false, af, false, b0, (short)0, a0, false, false);
    v16h b1 = load_bfrag_sw(wl, 4, lane, ki, 1);
    a1 = __builtin_amdgcn_wmma_f32_16x16x32_f16(false, af, false, b1, (short)0, a1, false, false);
    v16h b2 = load_bfrag_sw(wl, 4, lane, ki, 2);
    a2 = __builtin_amdgcn_wmma_f32_16x16x32_f16(false, af, false, b2, (short)0, a2, false, false);
    v16h b3 = load_bfrag_sw(wl, 4, lane, ki, 3);
    a3 = __builtin_amdgcn_wmma_f32_16x16x32_f16(false, af, false, b3, (short)0, a3, false, false);
  }
  int n0 = lane & 15;
  int mo = (lane & 16) ? 8 : 0;
#pragma unroll
  for (int r = 0; r < 8; r++) {
    _Float16* dr = act + (r + mo) * ACT_STRIDE;
    dr[n0]      = (_Float16)fmaxf(a0[r] + bias[n0],      0.f);
    dr[16 + n0] = (_Float16)fmaxf(a1[r] + bias[16 + n0], 0.f);
    dr[32 + n0] = (_Float16)fmaxf(a2[r] + bias[32 + n0], 0.f);
    dr[48 + n0] = (_Float16)fmaxf(a3[r] + bias[48 + n0], 0.f);
  }
}

// ---------------------------------------------------------------------------
// Fused 4-layer MLP: [K0,64,64,64,32], epilogue = mask + f16 out + scatter-add
// ---------------------------------------------------------------------------
struct MlpArgs {
  const _Float16* X;     // [M, K0] row-major
  int M, Kraw0, K0;
  const float *W0, *B0, *W1, *B1, *W2, *B2, *W3, *B3;   // f32 params
  const unsigned char* mask;   // per-row (mask_e) or nullptr
  _Float16* outH;              // [M, 32]
  float* scatR;                // [B*N*32] or nullptr
  const int* scatIdx;          // receiver node per row
  int Eper, Nn;                // rows-per-batch, node count
};

__global__ __launch_bounds__(128) void mlp4_kernel(MlpArgs a) {
  __shared__ __attribute__((aligned(32))) _Float16 wlds[128 * 64 + 64 * 64 + 64 * 64 + 64 * 32];
  __shared__ float blds[224];
  __shared__ __attribute__((aligned(32))) _Float16 act[MLP_WAVES][16 * ACT_STRIDE];

  int tid = threadIdx.x, lane = tid & 31, wave = tid >> 5;
  int K0 = a.K0;
  int w1 = K0 * 64, w2 = w1 + 4096, w3 = w2 + 4096;

  stage_w_sw(wlds,      a.W0, a.Kraw0, K0, 64, tid);
  stage_w_sw(wlds + w1, a.W1, 64, 64, 64, tid);
  stage_w_sw(wlds + w2, a.W2, 64, 64, 64, tid);
  stage_w_sw(wlds + w3, a.W3, 64, 64, 32, tid);
  for (int i = tid; i < 64; i += 128) { blds[i] = a.B0[i]; blds[64 + i] = a.B1[i]; blds[128 + i] = a.B2[i]; }
  for (int i = tid; i < 32; i += 128) blds[192 + i] = a.B3[i];
  __syncthreads();

  int row0 = blockIdx.x * (MLP_WAVES * 16) + wave * 16;
  _Float16* aw = &act[wave][0];

  // load 16-row input tile (16B vector chunks; K0 is a multiple of 32)
  {
    const _Float16* src = a.X + (size_t)row0 * K0;
    int cpr = K0 >> 3;                       // 16B chunks per row
    for (int ci = lane; ci < 16 * cpr; ci += 32) {
      int r = ci / cpr;
      int k = (ci - r * cpr) << 3;
      *(v8h*)(aw + r * ACT_STRIDE + k) = *(const v8h*)(src + (size_t)r * K0 + k);
    }
  }

  layer64(aw, wlds,      blds,       K0, lane);
  layer64(aw, wlds + w1, blds + 64,  64, lane);
  layer64(aw, wlds + w2, blds + 128, 64, lane);

  // final layer, C = 32
  v8f a0 = {}, a1 = {};
#pragma unroll
  for (int ki = 0; ki < 2; ki++) {
    v16h af = load_afrag(aw, lane, ki << 5);
    v16h b0 = load_bfrag_sw(wlds + w3, 2, lane, ki, 0);
    a0 = __builtin_amdgcn_wmma_f32_16x16x32_f16(false, af, false, b0, (short)0, a0, false, false);
    v16h b1 = load_bfrag_sw(wlds + w3, 2, lane, ki, 1);
    a1 = __builtin_amdgcn_wmma_f32_16x16x32_f16(false, af, false, b1, (short)0, a1, false, false);
  }
  int n0 = lane & 15;
  int mo = (lane & 16) ? 8 : 0;
#pragma unroll
  for (int r = 0; r < 8; r++) {
    int row = row0 + r + mo;
    bool on = a.mask ? (a.mask[row] != 0) : true;
    float v0 = on ? (a0[r] + blds[192 + n0]) : 0.f;
    float v1 = on ? (a1[r] + blds[208 + n0]) : 0.f;
    a.outH[(size_t)row * 32 + n0]      = (_Float16)v0;
    a.outH[(size_t)row * 32 + 16 + n0] = (_Float16)v1;
    if (a.scatR) {
      int b = row / a.Eper;
      int nd = a.scatIdx[row];
      float* base = a.scatR + ((size_t)b * a.Nn + nd) * 32;
      atomicAdd(base + n0, v0);
      atomicAdd(base + 16 + n0, v1);
    }
  }
}

// ---------------------------------------------------------------------------
// Edge index decode from one-hot Es/Er
// ---------------------------------------------------------------------------
__global__ void edge_idx_kernel(const float* Es, const float* Er, int* iS, int* iR,
                                int B, int N, int E) {
  int t = blockIdx.x * blockDim.x + threadIdx.x;
  if (t >= B * E) return;
  int b = t / E, e = t - b * E;
  const float* es = Es + (size_t)b * N * E + e;
  const float* er = Er + (size_t)b * N * E + e;
  int s = 0, r = 0;
  for (int n = 0; n < N; n++) {
    if (es[(size_t)n * E] != 0.f) s = n;
    if (er[(size_t)n * E] != 0.f) r = n;
  }
  iS[t] = s; iR[t] = r;
}

// ---------------------------------------------------------------------------
// Feature encoders (f16, stride-32 feature buffers)
// ---------------------------------------------------------------------------
__global__ void enc_nodes_kernel(const int* Va, _Float16* vv, int B, int N) {
  int t = blockIdx.x * blockDim.x + threadIdx.x;
  if (t >= B * N) return;
  int b = t / N, n = t - b * N;
  const int* base = Va + (size_t)b * 5 * N;
  int bt = base[0 * N + n] + 1;
  int nr = base[1 * N + n] + 1;
  float x = 2.f * (float)base[2 * N + n] / 50.f - 1.f;
  float y = 2.f * (float)base[3 * N + n] / 50.f - 1.f;
  int o6 = base[4 * N + n];
  _Float16* d = vv + (size_t)t * 32;
  for (int c = 0; c < 32; c++) d[c] = (_Float16)0.f;
  if (bt >= 0 && bt < 9) d[bt] = (_Float16)1.f;
  if (nr >= 0 && nr < 3) d[9 + nr] = (_Float16)1.f;
  d[12] = (_Float16)x;
  d[13] = (_Float16)y;
  if (o6 >= 0 && o6 < 6) d[14 + o6] = (_Float16)1.f;
}

__global__ void enc_edges_kernel(const int* Ea, _Float16* ve, int B, int E) {
  int t = blockIdx.x * blockDim.x + threadIdx.x;
  if (t >= B * E) return;
  int c0 = Ea[t];
  _Float16* d = ve + (size_t)t * 32;
  for (int c = 0; c < 32; c++) d[c] = (_Float16)0.f;
  if (c0 >= 0 && c0 < 4) d[c0] = (_Float16)1.f;
}

// ---------------------------------------------------------------------------
// Input assembly (concat + gather) into f16 X matrices
// ---------------------------------------------------------------------------
struct AsmEArgs {
  const _Float16* ve; int Ce;
  const _Float16* vv; int Cv;
  const float* u; int Cu;
  const int* iS; const int* iR;
  _Float16* X; int K0; int E; int N; int rows;
};
__global__ void asm_edge_kernel(AsmEArgs a) {
  int gid = blockIdx.x * blockDim.x + threadIdx.x;
  int row = gid / a.K0, k = gid - row * a.K0;
  if (row >= a.rows) return;
  _Float16 v = (_Float16)0.f;
  int c1 = a.Ce, c2 = a.Ce + a.Cv, c3 = a.Ce + 2 * a.Cv, c4 = c3 + a.Cu;
  if (k < c1) v = a.ve[(size_t)row * 32 + k];
  else if (k < c2) { int b = row / a.E; int s = a.iS[row]; v = a.vv[((size_t)b * a.N + s) * 32 + (k - c1)]; }
  else if (k < c3) { int b = row / a.E; int r = a.iR[row]; v = a.vv[((size_t)b * a.N + r) * 32 + (k - c2)]; }
  else if (k < c4) { int b = row / a.E; v = (_Float16)a.u[b * 32 + (k - c3)]; }
  a.X[(size_t)row * a.K0 + k] = v;
}

struct AsmVArgs {
  const _Float16* vv; int Cv;
  const float* R; int CR;
  const float* u; int Cu;
  _Float16* X; int K0; int N; int rows;
};
__global__ void asm_node_kernel(AsmVArgs a) {
  int gid = blockIdx.x * blockDim.x + threadIdx.x;
  int row = gid / a.K0, k = gid - row * a.K0;
  if (row >= a.rows) return;
  _Float16 v = (_Float16)0.f;
  int c1 = a.Cv, c2 = a.Cv + a.CR, c3 = c2 + a.Cu;
  if (k < c1) v = a.vv[(size_t)row * 32 + k];
  else if (k < c2) v = (_Float16)a.R[(size_t)row * 32 + (k - c1)];
  else if (k < c3) { int b = row / a.N; v = (_Float16)a.u[b * 32 + (k - c2)]; }
  a.X[(size_t)row * a.K0 + k] = v;
}

__global__ void fill_f32_kernel(float* p, int n) {
  int i = blockIdx.x * blockDim.x + threadIdx.x;
  if (i < n) p[i] = 0.f;
}

// ---------------------------------------------------------------------------
// Global MLP fu: P = [u, sum_V, sum_E] -> [Pin,64,64,64,32]
// ---------------------------------------------------------------------------
struct FuArgs {
  const _Float16* vv; int N;
  const _Float16* ve; int E;     // ve == nullptr => no edge term
  const float* uin; int Cu;
  const float *W0, *B0, *W1, *B1, *W2, *B2, *W3, *B3;
  float* uout;                   // [B,32]
};
__global__ __launch_bounds__(256) void fu_kernel(FuArgs a) {
  __shared__ float P[128], Ax[64], Ay[64], red[256];
  int b = blockIdx.x, tid = threadIdx.x;
  int c = tid & 31, part = tid >> 5;
  int Pin = a.Cu + 32 + (a.ve ? 32 : 0);

  float s = 0.f;
  for (int n = part; n < a.N; n += 8) s += (float)a.vv[((size_t)b * a.N + n) * 32 + c];
  red[tid] = s;
  __syncthreads();
  if (part == 0) { float t = 0.f; for (int p = 0; p < 8; p++) t += red[p * 32 + c]; P[a.Cu + c] = t; }
  __syncthreads();
  if (a.ve) {
    float s2 = 0.f;
    for (int e = part; e < a.E; e += 8) s2 += (float)a.ve[((size_t)b * a.E + e) * 32 + c];
    red[tid] = s2;
    __syncthreads();
    if (part == 0) { float t = 0.f; for (int p = 0; p < 8; p++) t += red[p * 32 + c]; P[a.Cu + 32 + c] = t; }
    __syncthreads();
  }
  if (tid < a.Cu) P[tid] = a.uin[b * 32 + tid];
  __syncthreads();

  if (tid < 64) { float ac = a.B0[tid]; for (int k = 0; k < Pin; k++) ac += P[k]  * a.W0[(size_t)k * 64 + tid]; Ax[tid] = fmaxf(ac, 0.f); }
  __syncthreads();
  if (tid < 64) { float ac = a.B1[tid]; for (int k = 0; k < 64;  k++) ac += Ax[k] * a.W1[(size_t)k * 64 + tid]; Ay[tid] = fmaxf(ac, 0.f); }
  __syncthreads();
  if (tid < 64) { float ac = a.B2[tid]; for (int k = 0; k < 64;  k++) ac += Ay[k] * a.W2[(size_t)k * 64 + tid]; Ax[tid] = fmaxf(ac, 0.f); }
  __syncthreads();
  if (tid < 32) { float ac = a.B3[tid]; for (int k = 0; k < 64;  k++) ac += Ax[k] * a.W3[(size_t)k * 32 + tid]; a.uout[b * 32 + tid] = ac; }
}

// ---------------------------------------------------------------------------
// Value head: val[b] = u[b,:] @ Wv + bv
// ---------------------------------------------------------------------------
__global__ void val_kernel(const float* u, const float* Wv, const float* bv, float* out, int B) {
  int b = threadIdx.x;
  if (b < B) {
    float s = bv[0];
    for (int cc = 0; cc < 32; cc++) s += u[b * 32 + cc] * Wv[cc];
    out[b] = s;
  }
}

// ---------------------------------------------------------------------------
// Final out-block (E=1) + masked softmax -> pol[b, 194]
// ---------------------------------------------------------------------------
struct PolArgs {
  const _Float16* vv; const float* u; int N;
  const float *W0, *B0, *W1, *B1, *W2, *B2, *W3, *B3;
  const unsigned char* mask;
  float* out;
};
__global__ __launch_bounds__(256) void pol_kernel(PolArgs a) {
  __shared__ float X[96], Ax[64], Ay[64], L[200];
  __shared__ float mx, sm;
  int b = blockIdx.x, tid = threadIdx.x;
  if (tid < 32) {
    X[tid]      = (float)a.vv[((size_t)b * a.N + (a.N - 1)) * 32 + tid];
    X[32 + tid] = 1.f;
    X[64 + tid] = a.u[b * 32 + tid];
  }
  __syncthreads();
  if (tid < 64) { float ac = a.B0[tid]; for (int k = 0; k < 96; k++) ac += X[k]  * a.W0[(size_t)k * 64 + tid]; Ax[tid] = fmaxf(ac, 0.f); }
  __syncthreads();
  if (tid < 64) { float ac = a.B1[tid]; for (int k = 0; k < 64; k++) ac += Ax[k] * a.W1[(size_t)k * 64 + tid]; Ay[tid] = fmaxf(ac, 0.f); }
  __syncthreads();
  if (tid < 64) { float ac = a.B2[tid]; for (int k = 0; k < 64; k++) ac += Ay[k] * a.W2[(size_t)k * 64 + tid]; Ax[tid] = fmaxf(ac, 0.f); }
  __syncthreads();
  if (tid < 194) {
    float ac = a.B3[tid];
    for (int k = 0; k < 64; k++) ac += Ax[k] * a.W3[(size_t)k * 194 + tid];
    if (!a.mask[b * 194 + tid]) ac -= 1e10f;
    L[tid] = ac;
  }
  __syncthreads();
  if (tid == 0) { float m = -3.0e38f; for (int i = 0; i < 194; i++) m = fmaxf(m, L[i]); mx = m; }
  __syncthreads();
  if (tid < 194) L[tid] = expf(L[tid] - mx);
  __syncthreads();
  if (tid == 0) { float t = 0.f; for (int i = 0; i < 194; i++) t += L[i]; sm = t; }
  __syncthreads();
  if (tid < 194) a.out[b * 194 + tid] = L[tid] / sm;
}

// ---------------------------------------------------------------------------
// Host driver
// ---------------------------------------------------------------------------
extern "C" void kernel_launch(void* const* d_in, const int* in_sizes, int n_in,
                              void* d_out, int out_size, void* d_ws, size_t ws_size,
                              hipStream_t stream) {
  (void)in_sizes; (void)n_in; (void)out_size; (void)ws_size;
  const int B = 4, N = 2048, E = 8192;

  const float* Es = (const float*)d_in[0];
  const float* Er = (const float*)d_in[1];
  const int*   Va = (const int*)d_in[172];
  const int*   Ea = (const int*)d_in[173];
  const unsigned char* mask_e = (const unsigned char*)d_in[174];
  const unsigned char* mask   = (const unsigned char*)d_in[176];
  auto PF = [&](int leaf) { return (const float*)d_in[2 + leaf]; };

  // workspace carve
  char* w = (char*)d_ws; size_t off = 0;
  auto carve = [&](size_t bytes) { off = (off + 255) & ~(size_t)255; void* p = w + off; off += bytes; return p; };
  int*      idxS = (int*)carve((size_t)B * E * 4);
  int*      idxR = (int*)carve((size_t)B * E * 4);
  _Float16* ve   = (_Float16*)carve((size_t)B * E * 32 * 2);
  _Float16* vv   = (_Float16*)carve((size_t)B * N * 32 * 2);
  _Float16* Xe   = (_Float16*)carve((size_t)B * E * 128 * 2);
  _Float16* Xv   = (_Float16*)carve((size_t)B * N * 128 * 2);
  float*    R    = (float*)carve((size_t)B * N * 32 * 4);
  float*    u    = (float*)carve((size_t)B * 32 * 4);

  edge_idx_kernel<<<(B * E + 255) / 256, 256, 0, stream>>>(Es, Er, idxS, idxR, B, N, E);

  auto run_branch = [&](const int blkBase[3], bool lastNoEdge) {
    enc_nodes_kernel<<<(B * N + 255) / 256, 256, 0, stream>>>(Va, vv, B, N);
    enc_edges_kernel<<<(B * E + 255) / 256, 256, 0, stream>>>(Ea, ve, B, E);
    int Ce = 4, Cv = 20, Cu = 0;
    for (int i = 0; i < 3; i++) {
      int fe = blkBase[i] + 0, fu = blkBase[i] + 8, fv = blkBase[i] + 16;
      bool hasEdge = !(lastNoEdge && i == 2);
      int CeOut = hasEdge ? 32 : 0;
      if (hasEdge) {
        int Kraw = Ce + 2 * Cv + Cu;
        int K0 = (Kraw + 31) / 32 * 32;
        int tot = B * E * K0;
        AsmEArgs ae{ve, Ce, vv, Cv, (Cu ? u : nullptr), Cu, idxS, idxR, Xe, K0, E, N, B * E};
        asm_edge_kernel<<<(tot + 255) / 256, 256, 0, stream>>>(ae);
        fill_f32_kernel<<<(B * N * 32 + 255) / 256, 256, 0, stream>>>(R, B * N * 32);
        MlpArgs me{Xe, B * E, Kraw, K0,
                   PF(fe + 0), PF(fe + 1), PF(fe + 2), PF(fe + 3),
                   PF(fe + 4), PF(fe + 5), PF(fe + 6), PF(fe + 7),
                   mask_e, ve, R, idxR, E, N};
        mlp4_kernel<<<(B * E) / 64, 128, 0, stream>>>(me);
      }
      {
        int Kraw = Cv + CeOut + Cu;
        int K0 = (Kraw + 31) / 32 * 32;
        int tot = B * N * K0;
        AsmVArgs av{vv, Cv, (CeOut ? R : nullptr), CeOut, (Cu ? u : nullptr), Cu, Xv, K0, N, B * N};
        asm_node_kernel<<<(tot + 255) / 256, 256, 0, stream>>>(av);
        MlpArgs mv{Xv, B * N, Kraw, K0,
                   PF(fv + 0), PF(fv + 1), PF(fv + 2), PF(fv + 3),
                   PF(fv + 4), PF(fv + 5), PF(fv + 6), PF(fv + 7),
                   nullptr, vv, nullptr, nullptr, N, N};
        mlp4_kernel<<<(B * N) / 64, 128, 0, stream>>>(mv);
      }
      {
        FuArgs fa{vv, N, (CeOut ? ve : nullptr), E, (Cu ? u : nullptr), Cu,
                  PF(fu + 0), PF(fu + 1), PF(fu + 2), PF(fu + 3),
                  PF(fu + 4), PF(fu + 5), PF(fu + 6), PF(fu + 7), u};
        fu_kernel<<<B, 256, 0, stream>>>(fa);
      }
      Ce = 32; Cv = 32; Cu = 32;
    }
  };

  // leaf bases: lin_val={0,1}; out.fe=2..9; pol blocks @26,50,74; val blocks @98,122,146
  const int valB[3] = {98, 122, 146};
  const int polB[3] = {26, 50, 74};

  run_branch(valB, false);
  val_kernel<<<1, 32, 0, stream>>>(u, PF(0), PF(1), (float*)d_out, B);

  run_branch(polB, true);
  PolArgs pa{vv, u, N,
             PF(2), PF(3), PF(4), PF(5), PF(6), PF(7), PF(8), PF(9),
             mask, ((float*)d_out) + B};
  pol_kernel<<<B, 256, 0, stream>>>(pa);
}
